// DGEBlock_4947802325309
// MI455X (gfx1250) — matrix operand order repaired and checked by reading, not compiled, not run on hardware
//
#include <hip/hip_runtime.h>
#include <hip/hip_bf16.h>

// ---------------------------------------------------------------------------
// Types for CDNA5 WMMA (wave32): v_wmma_f32_16x16x32_bf16
// ---------------------------------------------------------------------------
typedef __bf16 bf16;
typedef __bf16 v16bf __attribute__((ext_vector_type(16)));
typedef __bf16 v8bf  __attribute__((ext_vector_type(8)));
typedef float  v8f   __attribute__((ext_vector_type(8)));
typedef int    v4i   __attribute__((ext_vector_type(4)));
typedef int    v8i   __attribute__((ext_vector_type(8)));
typedef unsigned int v4u __attribute__((ext_vector_type(4)));

#define WMMA_BF16(A_, B_, C_) \
  __builtin_amdgcn_wmma_f32_16x16x32_bf16(false, (A_), false, (B_), (short)0, (C_), false, false)

static constexpr int Dm  = 2048;   // model dim
static constexpr int Hh  = 16;     // heads
static constexpr int HD  = 128;    // head dim
static constexpr int Bb  = 2;      // batch
static constexpr int Tt  = 2048;   // seq len
static constexpr int Mrows = Bb * Tt; // 4096 token rows

// ---------------------------------------------------------------------------
// CDNA5 async global->LDS copy (ASYNCcnt path). Signature confirmed by the
// round-2 diagnostic: (v4i32 AS1* src, v4i32 AS3* dst, imm offset, imm cpol).
// ---------------------------------------------------------------------------
#if defined(__HIP_DEVICE_COMPILE__) && defined(__gfx1250__) && \
    __has_builtin(__builtin_amdgcn_global_load_async_to_lds_b128)
#define HAVE_ASYNC_LDS 1
typedef __attribute__((address_space(1))) v4i as1_v4i;
typedef __attribute__((address_space(3))) v4i as3_v4i;
#else
#define HAVE_ASYNC_LDS 0
#endif

// Tensor Data Mover (TENSORcnt path). Round-4 diagnostic confirmed the 6-arg
// form: (uint32x4 g0, int32x8 g1, int32x4, int32x4, int32x8, i32 cpol).
#if defined(__HIP_DEVICE_COMPILE__) && defined(__gfx1250__) && \
    __has_builtin(__builtin_amdgcn_tensor_load_to_lds)
#define HAVE_TDM 1
typedef __attribute__((address_space(3))) void as3_void;
#else
#define HAVE_TDM 0
#endif

__device__ inline void async_cp16(void* lds, const void* g) {
#if HAVE_ASYNC_LDS
  __builtin_amdgcn_global_load_async_to_lds_b128((as1_v4i*)g, (as3_v4i*)lds, 0, 0);
#else
  *(v8bf*)lds = *(const v8bf*)g;
#endif
}

__device__ inline void async_wait0() {
#if HAVE_ASYNC_LDS
#if __has_builtin(__builtin_amdgcn_s_wait_asynccnt)
  __builtin_amdgcn_s_wait_asynccnt(0);
#else
  asm volatile("s_wait_asynccnt 0x0" ::: "memory");
#endif
#endif
}

#if HAVE_TDM
__device__ inline unsigned lds_addr_of(void* p) {
  return (unsigned)(unsigned long long)(as3_void*)p;
}

// One-shot TDM 2D tile load: `rows` x 32 bf16 tile from a row-major tensor with
// row stride K (elements) into LDS with a 4-DWORD pad after every 16 DWORDs
// (row stride 40 bf16 = LDA). D# layout per CDNA5 ISA §8.3/8.4.
__device__ inline void tdm_tile_2d(bf16* ldsdst, const bf16* gsrc, int rows, int K) {
  unsigned long long ga = (unsigned long long)(const void*)gsrc;
  v4u g0;
  g0[0] = 1u;                                   // count=1 (valid), user mode
  g0[1] = lds_addr_of(ldsdst);                  // LDS byte address
  g0[2] = (unsigned)ga;                         // global_addr[31:0]
  g0[3] = (unsigned)((ga >> 32) & 0x1ffffffu)   // global_addr[56:32]
        | (2u << 30);                           // type=2 ("image")
  unsigned td0 = (unsigned)K;                   // tensor_dim0 (elements)
  unsigned td1 = (unsigned)rows;                // tensor_dim1 (rows)
  v8i g1;
  g1[0] = (int)((1u << 16)    // data_size = 1 -> 2 bytes
              | (1u << 20)    // pad_enable
              | (3u << 22)    // pad_interval: 16 DWORDs (64B = tile row)
              | (3u << 25));  // pad_amount: 4 DWORDs (16B)
  g1[1] = (int)((td0 & 0xffffu) << 16);               // [63:48] = td0 lo
  g1[2] = (int)((td0 >> 16) | ((td1 & 0xffffu) << 16)); // td0 hi | td1 lo
  g1[3] = (int)((td1 >> 16) | (32u << 16));           // td1 hi | tile_dim0=32
  g1[4] = (int)td1;                                   // tile_dim1 = rows
  g1[5] = (int)td0;                                   // tensor_dim0_stride = K
  g1[6] = 0;
  g1[7] = 0;
  v4i z4 = {0, 0, 0, 0};
  v8i z8 = {0, 0, 0, 0, 0, 0, 0, 0};
  __builtin_amdgcn_tensor_load_to_lds(g0, g1, z4, z4, z8, 0);
}
#endif

__device__ inline v8f zero8() {
  v8f z;
  #pragma unroll
  for (int i = 0; i < 8; ++i) z[i] = 0.0f;
  return z;
}

// Build a 16-elem bf16 fragment from two 16-byte aligned pieces.
__device__ inline v16bf ld16(const bf16* p0, const bf16* p1) {
  union { v16bf v; v8bf h[2]; } u;
  u.h[0] = *(const v8bf*)p0;
  u.h[1] = *(const v8bf*)p1;
  return u.v;
}

// A-matrix 16x32 bf16 fragment (ISA 7.12.2): lane (lo,hi) holds row lo,
// K chunks [hi*8, hi*8+8) and [hi*8+16, hi*8+24).
__device__ inline v16bf load_a_frag(const bf16* base, int ld, int lane) {
  int lo = lane & 15, hi = (lane >> 4) & 1;
  const bf16* r = base + lo * ld + hi * 8;
  return ld16(r, r + 16);
}

// B-matrix 32x16 bf16 fragment: lane (lo,hi) holds column lo, K chunks [hi*8, hi*8+8) and [hi*8+16, hi*8+24).
__device__ inline v16bf load_b_frag(const bf16* base, int ld, int lane) {
  int lo = lane & 15, hi = (lane >> 4) & 1;
  const bf16* r = base + lo * ld + hi * 8;
  return ld16(r, r + 16);
}

__device__ inline float wave16_max(float x) {
  #pragma unroll
  for (int m = 1; m < 16; m <<= 1) x = fmaxf(x, __shfl_xor(x, m, 32));
  return x;
}
__device__ inline float wave16_add(float x) {
  #pragma unroll
  for (int m = 1; m < 16; m <<= 1) x += __shfl_xor(x, m, 32);
  return x;
}
__device__ inline float wave32_add(float x) {
  #pragma unroll
  for (int m = 1; m < 32; m <<= 1) x += __shfl_xor(x, m, 32);
  return x;
}

// ---------------------------------------------------------------------------
// f32 -> bf16 conversion (weights / activations)
// ---------------------------------------------------------------------------
__global__ __launch_bounds__(256) void cvt_f32_bf16(const float* __restrict__ in,
                                                    bf16* __restrict__ out, long n) {
  long i = (long)blockIdx.x * 256 + threadIdx.x;
  long stride = (long)gridDim.x * 256;
  for (; i < n; i += stride) out[i] = (bf16)in[i];
}

// ---------------------------------------------------------------------------
// LayerNorm: one block per token row (D = 2048), f32 in -> bf16 out
// ---------------------------------------------------------------------------
__global__ __launch_bounds__(256) void ln_kernel(const float* __restrict__ x,
                                                 const float* __restrict__ g,
                                                 const float* __restrict__ b,
                                                 bf16* __restrict__ out) {
  __shared__ float red[16];
  const int row = blockIdx.x;
  const int tid = threadIdx.x;
  const int wave = tid >> 5, lane = tid & 31;
  const float* xr = x + (size_t)row * Dm;

  float v[8];
  float s = 0.0f;
  #pragma unroll
  for (int i = 0; i < 8; ++i) { v[i] = xr[tid + i * 256]; s += v[i]; }
  s = wave32_add(s);
  if (lane == 0) red[wave] = s;
  __syncthreads();
  float tot = 0.0f;
  #pragma unroll
  for (int w = 0; w < 8; ++w) tot += red[w];
  const float mean = tot * (1.0f / Dm);

  float sq = 0.0f;
  #pragma unroll
  for (int i = 0; i < 8; ++i) { float d = v[i] - mean; sq += d * d; }
  sq = wave32_add(sq);
  __syncthreads();
  if (lane == 0) red[8 + wave] = sq;
  __syncthreads();
  float vt = 0.0f;
  #pragma unroll
  for (int w = 0; w < 8; ++w) vt += red[8 + w];
  const float rstd = rsqrtf(vt * (1.0f / Dm) + 1e-5f);

  bf16* orow = out + (size_t)row * Dm;
  #pragma unroll
  for (int i = 0; i < 8; ++i) {
    int c = tid + i * 256;
    orow[c] = (bf16)((v[i] - mean) * rstd * g[c] + b[c]);
  }
}

// ---------------------------------------------------------------------------
// Fused gated GEMM:  Y = (A @ W^T + b) * sigmoid(A @ Wg^T + bg)
//   A: [M,K] bf16 row-major, W/Wg: [N,K] bf16 row-major (K contiguous).
// mode 0: bf16 out; mode 1: bf16 out + exact-erf GELU; mode 2: f32 out + residual.
// Block tile 128x128, BK=32, double-buffered LDS staging via the TDM
// (tensor_load_to_lds, one instruction per tile, wave 0 only) with async-LDS /
// VGPR fallbacks. 8 waves in a 4x2 grid; wave tile 32x64.
// ---------------------------------------------------------------------------
static constexpr int LDA = 40; // 32 + 8 pad (bf16 elems, keeps 16B alignment)

__device__ inline void stage_tiles(const bf16* __restrict__ A,
                                   const bf16* __restrict__ W,
                                   const bf16* __restrict__ Wg,
                                   bf16* as, bf16* ws, bf16* wgs,
                                   int m0, int n0, int k0, int K, int tid) {
#if HAVE_TDM
  if (tid < 32) {  // wave 0 issues the three tile DMAs
    tdm_tile_2d(as, A + (size_t)m0 * K + k0, 128, K);
    tdm_tile_2d(ws, W + (size_t)n0 * K + k0, 128, K);
    tdm_tile_2d(wgs, Wg + (size_t)n0 * K + k0, 128, K);
  }
#else
  #pragma unroll
  for (int i = 0; i < 2; ++i) {
    int c = tid + i * 256;
    int row = c >> 2, kc = (c & 3) * 8;
    async_cp16(as + row * LDA + kc, A + (size_t)(m0 + row) * K + k0 + kc);
    async_cp16(ws + row * LDA + kc, W + (size_t)(n0 + row) * K + k0 + kc);
    async_cp16(wgs + row * LDA + kc, Wg + (size_t)(n0 + row) * K + k0 + kc);
  }
#endif
}

__device__ inline void stage_wait(int tid) {
#if HAVE_TDM
  if (tid < 32) {
#if __has_builtin(__builtin_amdgcn_s_wait_tensorcnt)
    __builtin_amdgcn_s_wait_tensorcnt(0);
#else
    asm volatile("s_wait_tensorcnt 0x0" ::: "memory");
#endif
  }
#else
  async_wait0();
#endif
}

__global__ __launch_bounds__(256) void gated_gemm(const bf16* __restrict__ A,
                                                  const bf16* __restrict__ W,
                                                  const float* __restrict__ bias,
                                                  const bf16* __restrict__ Wg,
                                                  const float* __restrict__ biasg,
                                                  const float* __restrict__ resid,
                                                  float* __restrict__ outf,
                                                  bf16* __restrict__ outb,
                                                  int N, int K, int mode) {
  __shared__ bf16 As[2][128 * LDA];
  __shared__ bf16 Ws_[2][128 * LDA];
  __shared__ bf16 Wgs[2][128 * LDA];

  const int tid = threadIdx.x;
  const int lane = tid & 31;
  const int wave = tid >> 5;
  const int m0 = blockIdx.y * 128;
  const int n0 = blockIdx.x * 128;
  const int r0 = (wave >> 1) * 32;  // wave rows within block tile
  const int c0 = (wave & 1) * 64;   // wave cols within block tile

  v8f acc[2][4], accg[2][4];
  #pragma unroll
  for (int i = 0; i < 2; ++i)
    #pragma unroll
    for (int j = 0; j < 4; ++j) { acc[i][j] = zero8(); accg[i][j] = zero8(); }

  int buf = 0;
  stage_tiles(A, W, Wg, As[0], Ws_[0], Wgs[0], m0, n0, 0, K, tid);

  for (int k0 = 0; k0 < K; k0 += 32) {
    stage_wait(tid);     // tile for `buf` has landed in LDS
    __syncthreads();     // and every wave is done with the other buffer
    if (k0 + 32 < K)
      stage_tiles(A, W, Wg, As[buf ^ 1], Ws_[buf ^ 1], Wgs[buf ^ 1],
                  m0, n0, k0 + 32, K, tid);

    const bf16* as = As[buf];
    const bf16* ws = Ws_[buf];
    const bf16* wgs = Wgs[buf];
    v16bf a0 = load_a_frag(as + (size_t)r0 * LDA, LDA, lane);
    v16bf a1 = load_a_frag(as + (size_t)(r0 + 16) * LDA, LDA, lane);
    #pragma unroll
    for (int ni = 0; ni < 4; ++ni) {
      v16bf bw = load_b_frag(ws + (size_t)(c0 + ni * 16) * LDA, LDA, lane);
      acc[0][ni] = WMMA_BF16(a0, bw, acc[0][ni]);
      acc[1][ni] = WMMA_BF16(a1, bw, acc[1][ni]);
      v16bf bg = load_b_frag(wgs + (size_t)(c0 + ni * 16) * LDA, LDA, lane);
      accg[0][ni] = WMMA_BF16(a0, bg, accg[0][ni]);
      accg[1][ni] = WMMA_BF16(a1, bg, accg[1][ni]);
    }
    buf ^= 1;
  }

  // Epilogue: bias + sigmoid gate (+gelu / +residual), C layout per ISA 7.12.2.
  const int lo = lane & 15, hi = (lane >> 4) & 1;
  #pragma unroll
  for (int mi = 0; mi < 2; ++mi) {
    #pragma unroll
    for (int ni = 0; ni < 4; ++ni) {
      v8f c = acc[mi][ni];
      v8f gg = accg[mi][ni];
      const int gcol = n0 + c0 + ni * 16 + lo;
      const float bb = bias[gcol];
      const float bg = biasg[gcol];
      #pragma unroll
      for (int j = 0; j < 8; ++j) {
        const int grow = m0 + r0 + mi * 16 + hi * 8 + j;
        float gate = gg[j] + bg;
        float y = (c[j] + bb) * (1.0f / (1.0f + __expf(-gate)));
        if (mode == 1) y = 0.5f * y * (1.0f + erff(y * 0.70710678118654752f));
        const size_t idx = (size_t)grow * N + gcol;
        if (mode == 2) outf[idx] = resid[idx] + y;
        else           outb[idx] = (bf16)y;
      }
    }
  }
}

// ---------------------------------------------------------------------------
// Flash attention (no mask): Q,K,V as bf16 [B,T,H,HD] laid out as [B*T, D] rows.
// Block: 256 threads = 8 waves; q-tile of 128 rows (16 per wave); K/V tiles of 64.
// K tile staged via async LDS copies; V transposed through VGPRs.
// ---------------------------------------------------------------------------
__global__ __launch_bounds__(256) void attention_kernel(const bf16* __restrict__ qb,
                                                        const bf16* __restrict__ kb,
                                                        const bf16* __restrict__ vb,
                                                        bf16* __restrict__ yb) {
  constexpr int LDK = 136; // 128 + 8 pad
  constexpr int LDV = 72;  // 64 + 8 pad
  constexpr int LDP = 72;
  __shared__ bf16 Ks[64 * LDK];        // K tile, rows = key, cols = head-dim
  __shared__ bf16 Vt[HD * LDV];        // V tile transposed: [head-dim][key]
  __shared__ bf16 Ps[8 * 16 * LDP];    // per-wave P staging (C->A relayout)

  const int tid = threadIdx.x;
  const int lane = tid & 31;
  const int wave = tid >> 5;
  const int lo = lane & 15, hi = (lane >> 4) & 1;

  const int qtile = blockIdx.x;        // 0..T/128-1
  const int bh = blockIdx.y;           // 0..B*H-1
  const int b = bh / Hh, h = bh % Hh;
  const size_t headoff = (size_t)b * Tt * Dm + (size_t)h * HD;
  const int qrow0 = qtile * 128 + wave * 16;

  // Q fragments: 16 rows x HD=128 -> 4 A-frags of 16x32.
  v16bf qf[4];
  {
    const bf16* qbase = qb + headoff + (size_t)qrow0 * Dm;
    #pragma unroll
    for (int d = 0; d < 4; ++d) qf[d] = load_a_frag(qbase + d * 32, Dm, lane);
  }

  float mrow[8], lrow[8];
  v8f o[8];
  #pragma unroll
  for (int j = 0; j < 8; ++j) { mrow[j] = -1e30f; lrow[j] = 0.0f; }
  #pragma unroll
  for (int f = 0; f < 8; ++f) o[f] = zero8();

  const float sc = 0.088388347648318447f; // 1/sqrt(128)
  bf16* pw = Ps + wave * 16 * LDP;

  for (int kt = 0; kt < Tt / 64; ++kt) {
    __syncthreads();
    // Stage K tile (64 x 128) async and transposed V tile through VGPRs.
    #pragma unroll
    for (int i = 0; i < 4; ++i) {
      int c = tid + i * 256;
      int r = c >> 4, kc = (c & 15) * 8;
      async_cp16(Ks + r * LDK + kc,
                 kb + headoff + (size_t)(kt * 64 + r) * Dm + kc);
      v8bf vv = *(const v8bf*)(vb + headoff + (size_t)(kt * 64 + r) * Dm + kc);
      #pragma unroll
      for (int e = 0; e < 8; ++e) Vt[(kc + e) * LDV + r] = vv[e];
    }
    async_wait0();
    __syncthreads();

    // S = Q @ K^T over this 64-key tile: 4 N-frags x 4 K-steps.
    v8f s[4];
    #pragma unroll
    for (int n = 0; n < 4; ++n) s[n] = zero8();
    #pragma unroll
    for (int d = 0; d < 4; ++d) {
      #pragma unroll
      for (int n = 0; n < 4; ++n)
        s[n] = WMMA_BF16(qf[d], load_b_frag(Ks + n * 16 * LDK + d * 32, LDK, lane), s[n]);
    }

    // Online softmax update.
    float rowmax[8];
    #pragma unroll
    for (int j = 0; j < 8; ++j) {
      float mx = s[0][j] * sc;
      #pragma unroll
      for (int n = 1; n < 4; ++n) mx = fmaxf(mx, s[n][j] * sc);
      rowmax[j] = wave16_max(mx);
    }
    float scale[8];
    #pragma unroll
    for (int j = 0; j < 8; ++j) {
      float nm = fmaxf(mrow[j], rowmax[j]);
      scale[j] = __expf(mrow[j] - nm);
      mrow[j] = nm;
    }
    float rsum[8];
    #pragma unroll
    for (int j = 0; j < 8; ++j) rsum[j] = 0.0f;
    #pragma unroll
    for (int n = 0; n < 4; ++n) {
      #pragma unroll
      for (int j = 0; j < 8; ++j) {
        float p = __expf(s[n][j] * sc - mrow[j]);
        s[n][j] = p;
        rsum[j] += p;
      }
    }
    #pragma unroll
    for (int j = 0; j < 8; ++j) {
      lrow[j] = lrow[j] * scale[j] + wave16_add(rsum[j]);
    }
    #pragma unroll
    for (int f = 0; f < 8; ++f)
      #pragma unroll
      for (int j = 0; j < 8; ++j) o[f][j] *= scale[j];

    // Relayout P (C layout) -> LDS row-major for A-fragment reload.
    #pragma unroll
    for (int n = 0; n < 4; ++n)
      #pragma unroll
      for (int j = 0; j < 8; ++j)
        pw[(hi * 8 + j) * LDP + n * 16 + lo] = (bf16)s[n][j];

    // O += P @ V : 2 K-steps of 32 keys x 8 head-dim frags.
    #pragma unroll
    for (int kk = 0; kk < 2; ++kk) {
      v16bf ap = load_a_frag(pw + kk * 32, LDP, lane);
      #pragma unroll
      for (int f = 0; f < 8; ++f)
        o[f] = WMMA_BF16(ap, load_b_frag(Vt + f * 16 * LDV + kk * 32, LDV, lane), o[f]);
    }
  }

  // Normalize and write y as bf16 [B*T, D] at head column block.
  float rl[8];
  #pragma unroll
  for (int j = 0; j < 8; ++j) rl[j] = 1.0f / lrow[j];
  #pragma unroll
  for (int f = 0; f < 8; ++f) {
    #pragma unroll
    for (int j = 0; j < 8; ++j) {
      int trow = qrow0 + hi * 8 + j;
      yb[(size_t)b * Tt * Dm + (size_t)trow * Dm + h * HD + f * 16 + lo] =
          (bf16)(o[f][j] * rl[j]);
    }
  }
}

// ---------------------------------------------------------------------------
// Host-side orchestration
// ---------------------------------------------------------------------------
extern "C" void kernel_launch(void* const* d_in, const int* in_sizes, int n_in,
                              void* d_out, int out_size, void* d_ws, size_t ws_size,
                              hipStream_t stream) {
  (void)in_sizes; (void)n_in; (void)out_size; (void)ws_size;

  const float* x     = (const float*)d_in[0];
  const float* W_q   = (const float*)d_in[1];
  const float* b_q   = (const float*)d_in[2];
  const float* Wg_q  = (const float*)d_in[3];
  const float* bg_q  = (const float*)d_in[4];
  const float* W_k   = (const float*)d_in[5];
  const float* b_k   = (const float*)d_in[6];
  const float* Wg_k  = (const float*)d_in[7];
  const float* bg_k  = (const float*)d_in[8];
  const float* W_v   = (const float*)d_in[9];
  const float* b_v   = (const float*)d_in[10];
  const float* Wg_v  = (const float*)d_in[11];
  const float* bg_v  = (const float*)d_in[12];
  const float* W_o   = (const float*)d_in[13];
  const float* b_o   = (const float*)d_in[14];
  const float* Wg_o  = (const float*)d_in[15];
  const float* bg_o  = (const float*)d_in[16];
  const float* W_in  = (const float*)d_in[17];
  const float* b_in  = (const float*)d_in[18];
  const float* Wg_in = (const float*)d_in[19];
  const float* bg_in = (const float*)d_in[20];
  const float* W_out = (const float*)d_in[21];
  const float* b_out = (const float*)d_in[22];
  const float* Wg_out= (const float*)d_in[23];
  const float* bg_out= (const float*)d_in[24];
  const float* ln1_g = (const float*)d_in[25];
  const float* ln1_b = (const float*)d_in[26];
  const float* ln2_g = (const float*)d_in[27];
  const float* ln2_b = (const float*)d_in[28];

  char* wsp = (char*)d_ws;
  size_t off = 0;
  auto alloc = [&](size_t bytes) -> void* {
    void* p = wsp + off;
    off = (off + bytes + 255) & ~(size_t)255;
    return p;
  };

  const size_t DD   = (size_t)Dm * Dm;      // 2048*2048
  const size_t DD4  = 4 * DD;               // 8192*2048
  const size_t ACT  = (size_t)Mrows * Dm;   // 4096*2048

  bf16* wq  = (bf16*)alloc(DD * 2);   bf16* wgq = (bf16*)alloc(DD * 2);
  bf16* wk  = (bf16*)alloc(DD * 2);   bf16* wgk = (bf16*)alloc(DD * 2);
  bf16* wv  = (bf16*)alloc(DD * 2);   bf16* wgv = (bf16*)alloc(DD * 2);
  bf16* wo  = (bf16*)alloc(DD * 2);   bf16* wgo = (bf16*)alloc(DD * 2);
  bf16* wi  = (bf16*)alloc(DD4 * 2);  bf16* wgi = (bf16*)alloc(DD4 * 2);
  bf16* wu  = (bf16*)alloc(DD4 * 2);  bf16* wgu = (bf16*)alloc(DD4 * 2);
  bf16* hbf = (bf16*)alloc(ACT * 2);
  bf16* qb  = (bf16*)alloc(ACT * 2);
  bf16* kbf = (bf16*)alloc(ACT * 2);
  bf16* vbf = (bf16*)alloc(ACT * 2);
  bf16* ybf = (bf16*)alloc(ACT * 2);
  float* x1 = (float*)alloc(ACT * 4);
  bf16* h2  = (bf16*)alloc(ACT * 2);
  bf16* mbf = (bf16*)alloc((size_t)Mrows * 4 * Dm * 2);

  dim3 blk(256);
  dim3 cgrid(2048);
  // Weight downcasts.
  cvt_f32_bf16<<<cgrid, blk, 0, stream>>>(W_q, wq, (long)DD);
  cvt_f32_bf16<<<cgrid, blk, 0, stream>>>(Wg_q, wgq, (long)DD);
  cvt_f32_bf16<<<cgrid, blk, 0, stream>>>(W_k, wk, (long)DD);
  cvt_f32_bf16<<<cgrid, blk, 0, stream>>>(Wg_k, wgk, (long)DD);
  cvt_f32_bf16<<<cgrid, blk, 0, stream>>>(W_v, wv, (long)DD);
  cvt_f32_bf16<<<cgrid, blk, 0, stream>>>(Wg_v, wgv, (long)DD);
  cvt_f32_bf16<<<cgrid, blk, 0, stream>>>(W_o, wo, (long)DD);
  cvt_f32_bf16<<<cgrid, blk, 0, stream>>>(Wg_o, wgo, (long)DD);
  cvt_f32_bf16<<<cgrid, blk, 0, stream>>>(W_in, wi, (long)DD4);
  cvt_f32_bf16<<<cgrid, blk, 0, stream>>>(Wg_in, wgi, (long)DD4);
  cvt_f32_bf16<<<cgrid, blk, 0, stream>>>(W_out, wu, (long)DD4);
  cvt_f32_bf16<<<cgrid, blk, 0, stream>>>(Wg_out, wgu, (long)DD4);

  // LN1
  ln_kernel<<<dim3(Mrows), blk, 0, stream>>>(x, ln1_g, ln1_b, hbf);

  // Q, K, V gated projections (bf16 out, mode 0).
  dim3 gemmDD(Dm / 128, Mrows / 128);
  gated_gemm<<<gemmDD, blk, 0, stream>>>(hbf, wq, b_q, wgq, bg_q,
                                         nullptr, nullptr, qb, Dm, Dm, 0);
  gated_gemm<<<gemmDD, blk, 0, stream>>>(hbf, wk, b_k, wgk, bg_k,
                                         nullptr, nullptr, kbf, Dm, Dm, 0);
  gated_gemm<<<gemmDD, blk, 0, stream>>>(hbf, wv, b_v, wgv, bg_v,
                                         nullptr, nullptr, vbf, Dm, Dm, 0);

  // Attention.
  attention_kernel<<<dim3(Tt / 128, Bb * Hh), blk, 0, stream>>>(qb, kbf, vbf, ybf);

  // Output projection + first residual (f32 out, mode 2).
  gated_gemm<<<gemmDD, blk, 0, stream>>>(ybf, wo, b_o, wgo, bg_o,
                                         x, x1, nullptr, Dm, Dm, 2);

  // LN2
  ln_kernel<<<dim3(Mrows), blk, 0, stream>>>(x1, ln2_g, ln2_b, h2);

  // MLP in: gated + exact GELU (mode 1), N = 8192.
  dim3 gemmIn(4 * Dm / 128, Mrows / 128);
  gated_gemm<<<gemmIn, blk, 0, stream>>>(h2, wi, b_in, wgi, bg_in,
                                         nullptr, nullptr, mbf, 4 * Dm, Dm, 1);

  // MLP out: gated + second residual into d_out (mode 2), K = 8192.
  gated_gemm<<<gemmDD, blk, 0, stream>>>(mbf, wu, b_out, wgu, bg_out,
                                         x1, (float*)d_out, nullptr, Dm, 4 * Dm, 2);
}
